// TinyGRU_18940805775601
// MI455X (gfx1250) — matrix operand order repaired and checked by reading, not compiled, not run on hardware
//
#include <hip/hip_runtime.h>

typedef __attribute__((ext_vector_type(16))) _Float16 v16h;
typedef __attribute__((ext_vector_type(8)))  float    v8f;

#define B_  512
#define T_  1024
#define D_  64
#define H_  128

#define HH_LD 136  // f16 h-tile row stride (272B = 68 dwords -> conflict-free column reads)
#define X_LD  68   // f16 x-tile row stride (136B = 34 dwords -> conflict-free column reads)

__device__ __forceinline__ v8f wmma_f16(v16h a, v16h b, v8f c) {
  // D = A(16x32 f16) * B(32x16 f16) + C(16x16 f32)
  return __builtin_amdgcn_wmma_f32_16x16x32_f16(false, a, false, b, (short)0, c, false, false);
}

// B-fragment (32x16, f16) from row-major f32 weight W[g][k]; B[k][n] = W[gbase+n][k].
__device__ __forceinline__ v16h load_bfrag_f32(const float* __restrict__ W, int ld,
                                               int gbase, int kbase, int lane) {
  const int n    = lane & 15;
  const int koff = (lane >> 4) << 3;
  const float* p = W + (size_t)(gbase + n) * ld + kbase + koff;
  v16h b;
#pragma unroll
  for (int j = 0; j < 16; ++j) b[j] = (_Float16)p[(j < 8) ? j : j + 8];
  return b;
}

// A-fragment (16x32, f16) from row-major f16 LDS tile.
__device__ __forceinline__ v16h afrag_from_h16(const _Float16* src, int ld, int kbase, int lane) {
  const int m = lane & 15;
  const int o = (lane >> 4) << 3;
  const _Float16* p = src + m * ld + kbase + o;
  v16h a;
#pragma unroll
  for (int i = 0; i < 8; ++i) { a[i] = p[i]; a[8 + i] = p[16 + i]; }
  return a;
}

// fast sigmoid/tanh: v_exp_f32 + v_rcp_f32, no IEEE-divide sequence
__device__ __forceinline__ float sigmf(float x) {
  return __builtin_amdgcn_rcpf(1.0f + __expf(-x));
}
__device__ __forceinline__ float tanh_fast(float x) {
  return __builtin_fmaf(2.0f, sigmf(2.0f * x), -1.0f);
}

__global__ __launch_bounds__(256, 1)
void tinygru_scan_kernel(const float* __restrict__ x,    const float* __restrict__ mask,
                         const float* __restrict__ w_ih, const float* __restrict__ w_hh,
                         const float* __restrict__ b_ih, const float* __restrict__ b_hh,
                         const float* __restrict__ Wo,   const float* __restrict__ bo,
                         float* __restrict__ out)
{
  __shared__ _Float16 h16_s[2][16 * HH_LD]; // double-buffered f16 hidden tile (WMMA input)
  __shared__ _Float16 x_s[2][16 * X_LD];    // double-buffered f16 x_t tile
  __shared__ float    mask_s[2][16];
  __shared__ float    wo_s[2 * H_];
  __shared__ float    bo_s[2];

  const int tid     = threadIdx.x;
  const int lane    = tid & 31;
  const int wv      = tid >> 5;            // 0..7, owns hidden cols [16*wv, 16*wv+16)
  const int rowbase = blockIdx.x * 16;     // 16 batch rows per block

  // ---- init LDS ----
  for (int i = tid; i < 16 * HH_LD; i += 256) h16_s[0][i] = (_Float16)0.0f;
  for (int i = tid; i < 2 * H_;     i += 256) wo_s[i] = Wo[i];
  if (tid < 2) bo_s[tid] = bo[tid];

  const int xrow = tid >> 4;               // 0..15
  const int xcol = (tid & 15) << 2;        // 0,4,...,60
  {
    const float4 v = *(const float4*)(x + ((size_t)(rowbase + xrow) * T_ + 0) * D_ + xcol);
    x_s[0][xrow * X_LD + xcol + 0] = (_Float16)v.x;
    x_s[0][xrow * X_LD + xcol + 1] = (_Float16)v.y;
    x_s[0][xrow * X_LD + xcol + 2] = (_Float16)v.z;
    x_s[0][xrow * X_LD + xcol + 3] = (_Float16)v.w;
    if (tid < 16) mask_s[0][tid] = mask[(size_t)(rowbase + tid) * T_ + 0];
  }

  // ---- hoist loop-invariant weight B-fragments into VGPRs (f16) ----
  const int gb[3] = { wv * 16, H_ + wv * 16, 2 * H_ + wv * 16 };  // r, z, n column tiles
  v16h bih[3][2], bhh[3][4];
#pragma unroll
  for (int g = 0; g < 3; ++g) {
#pragma unroll
    for (int kb = 0; kb < 2; ++kb) bih[g][kb] = load_bfrag_f32(w_ih, D_, gb[g], kb * 32, lane);
#pragma unroll
    for (int kb = 0; kb < 4; ++kb) bhh[g][kb] = load_bfrag_f32(w_hh, H_, gb[g], kb * 32, lane);
  }

  // ---- per-lane constants (C/D layout: N = lane&15, M = rr + 8*(lane>>4)) ----
  const int ncol = lane & 15;
  const int hi   = lane >> 4;
  const int col  = wv * 16 + ncol;                       // hidden column 0..127
  const float c_br  = b_ih[col]          + b_hh[col];
  const float c_bz  = b_ih[H_ + col]     + b_hh[H_ + col];
  const float c_bxn = b_ih[2 * H_ + col];
  const float c_bhn = b_hh[2 * H_ + col];

  // bias vectors as WMMA C-operand seeds: bias depends only on this lane's column,
  // so broadcasting over the 8 row-slots reproduces the bias matrix exactly.
  v8f cr0, cz0, cxn0, chn0;
#pragma unroll
  for (int i = 0; i < 8; ++i) { cr0[i] = c_br; cz0[i] = c_bz; cxn0[i] = c_bxn; chn0[i] = c_bhn; }

  // f32 master copy of this lane's h elements: rows (hi*8 + rr), column `col`
  float h_reg[8];
#pragma unroll
  for (int rr = 0; rr < 8; ++rr) h_reg[rr] = 0.0f;

  // logit phase: 32 groups of 8 lanes; group = (row, output); lane handles 16 contiguous cols
  const int g8 = tid >> 3, l8 = tid & 7;
  const int lrow = g8 >> 1, lo = g8 & 1;
  const int lchunk = l8 * 16;

  __syncthreads();

  for (int t = 0; t < T_; ++t) {
    const int cur = t & 1, nxt = cur ^ 1;

    // -- prefetch x_{t+1}, mask_{t+1} into registers --
    float4 xnext = {0.f, 0.f, 0.f, 0.f};
    float  mnext = 0.f;
    if (t + 1 < T_) {
      xnext = *(const float4*)(x + ((size_t)(rowbase + xrow) * T_ + (t + 1)) * D_ + xcol);
      if (tid < 16) mnext = mask[(size_t)(rowbase + tid) * T_ + (t + 1)];
    }

    // -- A fragments (pure f16 DS loads) --
    v16h ax0 = afrag_from_h16(&x_s[cur][0],   X_LD,  0,  lane);
    v16h ax1 = afrag_from_h16(&x_s[cur][0],   X_LD,  32, lane);
    v16h ah0 = afrag_from_h16(&h16_s[cur][0], HH_LD, 0,  lane);
    v16h ah1 = afrag_from_h16(&h16_s[cur][0], HH_LD, 32, lane);
    v16h ah2 = afrag_from_h16(&h16_s[cur][0], HH_LD, 64, lane);
    v16h ah3 = afrag_from_h16(&h16_s[cur][0], HH_LD, 96, lane);

    // -- 18 WMMAs; accumulators seeded with folded biases (free bias add) --
    v8f ar = cr0, az = cz0, axn = cxn0, ahn = chn0;
    ar  = wmma_f16(ax0, bih[0][0], ar);   ar  = wmma_f16(ax1, bih[0][1], ar);
    ar  = wmma_f16(ah0, bhh[0][0], ar);   ar  = wmma_f16(ah1, bhh[0][1], ar);
    ar  = wmma_f16(ah2, bhh[0][2], ar);   ar  = wmma_f16(ah3, bhh[0][3], ar);

    az  = wmma_f16(ax0, bih[1][0], az);   az  = wmma_f16(ax1, bih[1][1], az);
    az  = wmma_f16(ah0, bhh[1][0], az);   az  = wmma_f16(ah1, bhh[1][1], az);
    az  = wmma_f16(ah2, bhh[1][2], az);   az  = wmma_f16(ah3, bhh[1][3], az);

    axn = wmma_f16(ax0, bih[2][0], axn);  axn = wmma_f16(ax1, bih[2][1], axn);

    ahn = wmma_f16(ah0, bhh[2][0], ahn);  ahn = wmma_f16(ah1, bhh[2][1], ahn);
    ahn = wmma_f16(ah2, bhh[2][2], ahn);  ahn = wmma_f16(ah3, bhh[2][3], ahn);

    // -- gate nonlinearities + masked update; h_old lives in registers --
#pragma unroll
    for (int rr = 0; rr < 8; ++rr) {
      const int   mrow = rr + hi * 8;
      const float hold = h_reg[rr];
      const float r  = sigmf(ar[rr]);
      const float z  = sigmf(az[rr]);
      const float nn = tanh_fast(__builtin_fmaf(r, ahn[rr], axn[rr]));
      const float hc  = __builtin_fmaf(z, hold - nn, nn);          // (1-z)*n + z*h
      const float mm  = mask_s[cur][mrow];
      const float hn2 = __builtin_fmaf(mm, hc - hold, hold);       // m*hc + (1-m)*h
      h_reg[rr] = hn2;
      h16_s[nxt][mrow * HH_LD + col] = (_Float16)hn2;
    }

    if (t + 1 < T_) {
      x_s[nxt][xrow * X_LD + xcol + 0] = (_Float16)xnext.x;
      x_s[nxt][xrow * X_LD + xcol + 1] = (_Float16)xnext.y;
      x_s[nxt][xrow * X_LD + xcol + 2] = (_Float16)xnext.z;
      x_s[nxt][xrow * X_LD + xcol + 3] = (_Float16)xnext.w;
      if (tid < 16) mask_s[nxt][tid] = mnext;
    }

    __syncthreads();   // publish h16_s[nxt], x_s[nxt], mask_s[nxt]; single barrier per step

    // -- logits: each lane reads a contiguous 16-half chunk (vectorizes to ds_load_b128) --
    {
      const _Float16* hp = &h16_s[nxt][lrow * HH_LD + lchunk];
      const float*    wp = &wo_s[lo * H_ + lchunk];
      float p = 0.0f;
#pragma unroll
      for (int j = 0; j < 16; ++j) p = __builtin_fmaf((float)hp[j], wp[j], p);
      p += __shfl_down(p, 4, 8);
      p += __shfl_down(p, 2, 8);
      p += __shfl_down(p, 1, 8);
      if (l8 == 0)
        out[(((size_t)(rowbase + lrow)) * T_ + t) * 2 + lo] = p + bo_s[lo];
    }
  }
}

extern "C" void kernel_launch(void* const* d_in, const int* in_sizes, int n_in,
                              void* d_out, int out_size, void* d_ws, size_t ws_size,
                              hipStream_t stream) {
  const float* x    = (const float*)d_in[0];
  const float* mask = (const float*)d_in[1];
  const float* w_ih = (const float*)d_in[2];
  const float* w_hh = (const float*)d_in[3];
  const float* b_ih = (const float*)d_in[4];
  const float* b_hh = (const float*)d_in[5];
  const float* Wo   = (const float*)d_in[6];
  const float* bo   = (const float*)d_in[7];
  float* out = (float*)d_out;

  dim3 grid(B_ / 16);   // 32 persistent blocks, one 16-row batch tile each
  dim3 block(256);      // 8 wave32s
  tinygru_scan_kernel<<<grid, block, 0, stream>>>(x, mask, w_ih, w_hh, b_ih, b_hh, Wo, bo, out);
}
